// RankAugmentedLinearAttention_59725815218632
// MI455X (gfx1250) — matrix operand order repaired and checked by reading, not compiled, not run on hardware
//
#include <hip/hip_runtime.h>

#define Bb 4
#define Ss 4096
#define Ee 1024
#define Hh 16
#define Dd 64
#define BS (Bb*Ss)

typedef __attribute__((ext_vector_type(16))) __bf16 v16bf;
typedef __attribute__((ext_vector_type(2)))  __bf16 v2bf;
typedef __attribute__((ext_vector_type(8)))  float  v8f;

union Ub {
    v16bf v;
    unsigned short us[16];
    unsigned u32[8];
    uint4 q[2];
};

static __device__ inline unsigned short f2bfu(float f) {
    unsigned u = __float_as_uint(f);
    return (unsigned short)((u + 0x7FFFu + ((u >> 16) & 1u)) >> 16);
}
// Packed 2x f32 -> 2x bf16 (one VALU op if HW cvt available)
static __device__ inline unsigned f2bf2(float a, float b) {
#if __has_builtin(__builtin_amdgcn_cvt_pk_bf16_f32)
    v2bf r = __builtin_amdgcn_cvt_pk_bf16_f32(a, b);
    return __builtin_bit_cast(unsigned, r);
#else
    return (unsigned)f2bfu(a) | ((unsigned)f2bfu(b) << 16);
#endif
}
static __device__ inline float bfu2f(unsigned short h) {
    return __uint_as_float(((unsigned)h) << 16);
}
static __device__ inline uint4 pack8(const float* v) {
    uint4 r;
    r.x = f2bf2(v[0], v[1]);
    r.y = f2bf2(v[2], v[3]);
    r.z = f2bf2(v[4], v[5]);
    r.w = f2bf2(v[6], v[7]);
    return r;
}
static __device__ inline v8f wmma_bf16(const Ub& a, const Ub& b, v8f c) {
    return __builtin_amdgcn_wmma_f32_16x16x32_bf16(false, a.v, false, b.v,
                                                   (short)0, c, false, false);
}

// ---------------------------------------------------------------------------
// Kernel 1: fused QKV projection.  Computes C^T tiles of  X @ W^T + b.
//   z=0: q_aug = relu(.)+eps       -> stored (B,S,E) row-major bf16
//   z=1: k_aug = mask? 0: relu+eps -> stored (B,H,D,S) transposed bf16
//   z=2: v                         -> stored (B,H,D,S) transposed bf16
// grid (E/64, BS/128, 3), block 256 (8 waves, 16 s-rows each).
// ---------------------------------------------------------------------------
__global__ __launch_bounds__(256) void proj_kernel(
    const float* __restrict__ qin, const float* __restrict__ kin,
    const float* __restrict__ vin,
    const float* __restrict__ Wq, const float* __restrict__ Wk,
    const float* __restrict__ Wv,
    const float* __restrict__ bq, const float* __restrict__ bk,
    const float* __restrict__ bv,
    const unsigned char* __restrict__ mask,
    unsigned short* __restrict__ qa, unsigned short* __restrict__ kt,
    unsigned short* __restrict__ vt)
{
    const int z = blockIdx.z;
    const float* X    = (z == 0) ? qin : (z == 1) ? kin : vin;
    const float* W    = (z == 0) ? Wq  : (z == 1) ? Wk  : Wv;
    const float* bias = (z == 0) ? bq  : (z == 1) ? bk  : bv;
    const int e0 = blockIdx.x * 64;
    const int tid = threadIdx.x;

    __shared__ __align__(16) unsigned short sW[64 * 1032]; // 64 feat rows, padded

    // Stage 64x1024 f32 weight slab -> bf16 LDS
    for (int i = tid; i < 64 * 256; i += 256) {
        int row = i >> 8, c4 = (i & 255) * 4;
        float4 w = *(const float4*)(W + (size_t)(e0 + row) * Ee + c4);
        unsigned* d = (unsigned*)&sW[row * 1032 + c4];
        d[0] = f2bf2(w.x, w.y);
        d[1] = f2bf2(w.z, w.w);
    }
    __syncthreads();

    const int wid = tid >> 5, lane = tid & 31;
    const int lq = lane & 15, hf = lane >> 4;
    const int srow = blockIdx.y * 128 + wid * 16 + lq;

    v8f acc[4] = {};
    for (int kk = 0; kk < Ee; kk += 32) {
        // B operand: 16 K-contiguous f32 from X row (this lane's s-row)
        Ub bfr;
        const float* xp = X + (size_t)srow * Ee + kk + hf * 16;
#pragma unroll
        for (int j = 0; j < 4; j++) {
            float4 x = ((const float4*)xp)[j];
            bfr.u32[2 * j]     = f2bf2(x.x, x.y);
            bfr.u32[2 * j + 1] = f2bf2(x.z, x.w);
        }
#pragma unroll
        for (int t = 0; t < 4; t++) {
            Ub af;
            const unsigned short* ap = &sW[(t * 16 + lq) * 1032 + kk + hf * 16];
            af.q[0] = *(const uint4*)ap;
            af.q[1] = *(const uint4*)(ap + 8);
            acc[t] = wmma_bf16(af, bfr, acc[t]);
        }
    }

    // Epilogue: bias, activation, mask, store
    const int b = srow >> 12, s = srow & (Ss - 1);
    const bool mk = (z == 1) && (mask[srow] != 0);
#pragma unroll
    for (int t = 0; t < 4; t++) {
        float vals[8];
        const int ebase = e0 + t * 16 + hf * 8;
#pragma unroll
        for (int r = 0; r < 8; r++) {
            float v = acc[t][r] + bias[ebase + r];
            if (z < 2) v = fmaxf(v, 0.0f) + 1e-6f;
            if (mk) v = 0.0f;
            vals[r] = v;
        }
        if (z == 0) {
            *(uint4*)(qa + (size_t)srow * Ee + ebase) = pack8(vals);
        } else {
            unsigned short* dst = (z == 1) ? kt : vt;
#pragma unroll
            for (int r = 0; r < 8; r++) {
                int e = ebase + r, h = e >> 6, d = e & 63;
                dst[(((size_t)b * Hh + h) * Dd + d) * Ss + s] = f2bfu(vals[r]);
            }
        }
    }
}

// ---------------------------------------------------------------------------
// Kernel 2: per (b,h)  kv[d,v] = sum_s k_aug[s,d] * v[s,v]  and  k_sum[d].
// Both operands read K(=s)-contiguous from the transposed layouts.
// 8 waves: (2 d-halves) x (4 s-quarters); f32 reduction in LDS (ds_add_f32).
// kv stored transposed (v,d) bf16 so it is an A operand for the attn kernel.
// grid (B*H), block 256.
// ---------------------------------------------------------------------------
__global__ __launch_bounds__(256) void kv_kernel(
    const unsigned short* __restrict__ kt, const unsigned short* __restrict__ vt,
    unsigned short* __restrict__ kvt, float* __restrict__ ksum)
{
    const int bh = blockIdx.x;
    const int tid = threadIdx.x;
    __shared__ float skv[64 * 65];
    __shared__ float sks[64];
    for (int i = tid; i < 64 * 65; i += 256) skv[i] = 0.0f;
    if (tid < 64) sks[tid] = 0.0f;
    __syncthreads();

    const int wid = tid >> 5, lane = tid & 31;
    const int lq = lane & 15, hf = lane >> 4;
    const int sq = wid & 3, dh = wid >> 2;
    const unsigned short* kbase = kt + (size_t)bh * 64 * Ss;
    const unsigned short* vbase = vt + (size_t)bh * 64 * Ss;

    v8f acc[2][4] = {};
    for (int kk = sq * 1024; kk < sq * 1024 + 1024; kk += 32) {
        Ub bfr[4];
#pragma unroll
        for (int u = 0; u < 4; u++) {
            const unsigned short* p = vbase + (size_t)(u * 16 + lq) * Ss + kk + hf * 16;
            bfr[u].q[0] = ((const uint4*)p)[0];
            bfr[u].q[1] = ((const uint4*)p)[1];
        }
#pragma unroll
        for (int td = 0; td < 2; td++) {
            Ub af;
            const int d = (dh * 2 + td) * 16 + lq;
            const unsigned short* p = kbase + (size_t)d * Ss + kk + hf * 16;
            af.q[0] = *(const uint4*)p;
            af.q[1] = *(const uint4*)(p + 8);
#pragma unroll
            for (int u = 0; u < 4; u++)
                acc[td][u] = wmma_bf16(af, bfr[u], acc[td][u]);
        }
    }

    // cross-wave f32 reduce in LDS
#pragma unroll
    for (int td = 0; td < 2; td++)
#pragma unroll
        for (int u = 0; u < 4; u++)
#pragma unroll
            for (int r = 0; r < 8; r++) {
                const int d = (dh * 2 + td) * 16 + r + hf * 8;
                const int v = u * 16 + lq;
                atomicAdd(&skv[d * 65 + v], acc[td][u][r]);
            }

    // k_sum: thread = (row, quarter of S)
    {
        const int row = tid & 63, quar = tid >> 6;
        const unsigned short* p = kbase + (size_t)row * Ss + quar * 1024;
        float s = 0.0f;
        for (int i = 0; i < 1024; i += 8) {
            uint4 qv = *(const uint4*)(p + i);
            const unsigned short* us = (const unsigned short*)&qv;
#pragma unroll
            for (int j = 0; j < 8; j++) s += bfu2f(us[j]);
        }
        atomicAdd(&sks[row], s);
    }
    __syncthreads();

    for (int i = tid; i < 4096; i += 256) {
        const int v = i >> 6, d = i & 63;
        kvt[(size_t)bh * 4096 + v * 64 + d] = f2bfu(skv[d * 65 + v]);
    }
    if (tid < 64) ksum[bh * 64 + tid] = sks[tid];
}

// ---------------------------------------------------------------------------
// Kernel 3: attn^T[v,q] = sum_d kvT[v,d] * q_aug[q,d]; divide by denom;
// store attn (B,S,E) row-major bf16.  grid (S/128, B*H), block 256.
// ---------------------------------------------------------------------------
__global__ __launch_bounds__(256) void attn_kernel(
    const unsigned short* __restrict__ qa, const unsigned short* __restrict__ kvt,
    const float* __restrict__ ksum, unsigned short* __restrict__ attn)
{
    const int bh = blockIdx.y;
    const int b = bh >> 4, h = bh & 15;
    const int tid = threadIdx.x;
    __shared__ float sks[64];
    if (tid < 64) sks[tid] = ksum[bh * 64 + tid];
    __syncthreads();

    const int wid = tid >> 5, lane = tid & 31;
    const int lq = lane & 15, hf = lane >> 4;
    const int q = blockIdx.x * 128 + wid * 16 + lq;
    const size_t qrow = ((size_t)b * Ss + q) * Ee + (size_t)h * 64;

    v8f acc[4] = {};
#pragma unroll
    for (int kk = 0; kk < 64; kk += 32) {
        Ub bfr;
        const unsigned short* p = qa + qrow + kk + hf * 16;
        bfr.q[0] = ((const uint4*)p)[0];
        bfr.q[1] = ((const uint4*)p)[1];
#pragma unroll
        for (int t = 0; t < 4; t++) {
            Ub af;
            const unsigned short* ap =
                kvt + (size_t)bh * 4096 + (t * 16 + lq) * 64 + kk + hf * 16;
            af.q[0] = *(const uint4*)ap;
            af.q[1] = *(const uint4*)(ap + 8);
            acc[t] = wmma_bf16(af, bfr, acc[t]);
        }
    }

    // denom = dot(q_aug[q,:], k_sum)
    float den = 0.0f;
#pragma unroll
    for (int d0 = 0; d0 < 64; d0 += 8) {
        uint4 qv = *(const uint4*)(qa + qrow + d0);
        const unsigned short* us = (const unsigned short*)&qv;
#pragma unroll
        for (int j = 0; j < 8; j++) den += bfu2f(us[j]) * sks[d0 + j];
    }
    den = fmaxf(den, 1e-6f);
    const float inv = 1.0f / den;

#pragma unroll
    for (int t = 0; t < 4; t++) {
        float vals[8];
#pragma unroll
        for (int r = 0; r < 8; r++) vals[r] = acc[t][r] * inv;
        *(uint4*)(attn + qrow + t * 16 + hf * 8) = pack8(vals);
    }
}

// ---------------------------------------------------------------------------
// Kernel 4: out^T[e,q] = sum_v Wo[e,v] * attn[q,v]; + bo; f32 out (d_out).
// grid (E/64, BS/128), block 256.
// ---------------------------------------------------------------------------
__global__ __launch_bounds__(256) void out_kernel(
    const unsigned short* __restrict__ attn, const float* __restrict__ Wo,
    const float* __restrict__ bo, float* __restrict__ out)
{
    const int e0 = blockIdx.x * 64;
    const int tid = threadIdx.x;
    __shared__ __align__(16) unsigned short sW[64 * 1032];
    for (int i = tid; i < 64 * 256; i += 256) {
        int row = i >> 8, c4 = (i & 255) * 4;
        float4 w = *(const float4*)(Wo + (size_t)(e0 + row) * Ee + c4);
        unsigned* d = (unsigned*)&sW[row * 1032 + c4];
        d[0] = f2bf2(w.x, w.y);
        d[1] = f2bf2(w.z, w.w);
    }
    __syncthreads();

    const int wid = tid >> 5, lane = tid & 31;
    const int lq = lane & 15, hf = lane >> 4;
    const int srow = blockIdx.y * 128 + wid * 16 + lq;

    v8f acc[4] = {};
    for (int kk = 0; kk < Ee; kk += 32) {
        Ub bfr;
        const unsigned short* p = attn + (size_t)srow * Ee + kk + hf * 16;
        bfr.q[0] = ((const uint4*)p)[0];
        bfr.q[1] = ((const uint4*)p)[1];
#pragma unroll
        for (int t = 0; t < 4; t++) {
            Ub af;
            const unsigned short* ap = &sW[(t * 16 + lq) * 1032 + kk + hf * 16];
            af.q[0] = *(const uint4*)ap;
            af.q[1] = *(const uint4*)(ap + 8);
            acc[t] = wmma_bf16(af, bfr, acc[t]);
        }
    }

#pragma unroll
    for (int t = 0; t < 4; t++) {
        const int ebase = e0 + t * 16 + hf * 8;
        float4 o0, o1;
        o0.x = acc[t][0] + bo[ebase + 0];
        o0.y = acc[t][1] + bo[ebase + 1];
        o0.z = acc[t][2] + bo[ebase + 2];
        o0.w = acc[t][3] + bo[ebase + 3];
        o1.x = acc[t][4] + bo[ebase + 4];
        o1.y = acc[t][5] + bo[ebase + 5];
        o1.z = acc[t][6] + bo[ebase + 6];
        o1.w = acc[t][7] + bo[ebase + 7];
        *(float4*)(out + (size_t)srow * Ee + ebase)     = o0;
        *(float4*)(out + (size_t)srow * Ee + ebase + 4) = o1;
    }
}

// ---------------------------------------------------------------------------
extern "C" void kernel_launch(void* const* d_in, const int* in_sizes, int n_in,
                              void* d_out, int out_size, void* d_ws, size_t ws_size,
                              hipStream_t stream)
{
    (void)in_sizes; (void)n_in; (void)out_size; (void)ws_size;
    const float* query = (const float*)d_in[0];
    const float* key_  = (const float*)d_in[1];
    const float* value = (const float*)d_in[2];
    const unsigned char* mask = (const unsigned char*)d_in[3];
    const float* Wq = (const float*)d_in[4];
    const float* bq = (const float*)d_in[5];
    const float* Wk = (const float*)d_in[6];
    const float* bk = (const float*)d_in[7];
    const float* Wv = (const float*)d_in[8];
    const float* bv = (const float*)d_in[9];
    const float* Wo = (const float*)d_in[10];
    const float* bo = (const float*)d_in[11];

    char* ws = (char*)d_ws;
    const size_t MB = 1u << 20;
    unsigned short* qa  = (unsigned short*)(ws);             // 32 MB (B,S,E) bf16
    unsigned short* kt  = (unsigned short*)(ws + 32 * MB);   // 32 MB (B,H,D,S) bf16
    unsigned short* vt  = (unsigned short*)(ws + 64 * MB);   // 32 MB (B,H,D,S) bf16
    unsigned short* kvt = (unsigned short*)(ws + 96 * MB);   // 512 KB (B,H,V,D) bf16
    float*          ksum = (float*)(ws + 97 * MB);           // 16 KB (B,H,D) f32
    unsigned short* attn = kt;  // kt is dead after kv_kernel; reuse for attn

    proj_kernel<<<dim3(Ee / 64, BS / 128, 3), 256, 0, stream>>>(
        query, key_, value, Wq, Wk, Wv, bq, bk, bv, mask, qa, kt, vt);
    kv_kernel<<<dim3(Bb * Hh), 256, 0, stream>>>(kt, vt, kvt, ksum);
    attn_kernel<<<dim3(Ss / 128, Bb * Hh), 256, 0, stream>>>(qa, kvt, ksum, attn);
    out_kernel<<<dim3(Ee / 64, BS / 128), 256, 0, stream>>>(attn, Wo, bo, (float*)d_out);
}
